// TransformerLayer_28174985462169
// MI455X (gfx1250) — matrix operand order repaired and checked by reading, not compiled, not run on hardware
//
#include <hip/hip_runtime.h>
#include <hip/hip_bf16.h>

// ---------------------------------------------------------------------------
// Transformer-XL layer for MI455X (gfx1250, wave32, WMMA bf16).
// - every WMMA operand is a contiguous b128-loadable run (ISA 7.12.2 maps)
// - bf16 A tiles staged with GLOBAL_LOAD_ASYNC_TO_LDS_B128 (ASYNCcnt)
// ---------------------------------------------------------------------------

typedef __bf16 bf16_t;
typedef __attribute__((ext_vector_type(16))) __bf16 v16bf;
typedef __attribute__((ext_vector_type(8)))  __bf16 v8bf;
typedef __attribute__((ext_vector_type(4)))  __bf16 v4bf;
typedef __attribute__((ext_vector_type(8)))  float  v8f;
typedef __attribute__((ext_vector_type(4)))  float  v4f;

constexpr int B_  = 8;
constexpr int S_  = 512;
constexpr int M_  = 512;
constexpr int H_  = 1024;
constexpr int NH_ = 16;
constexpr int FF_ = 4096;
constexpr int DH_ = 64;
constexpr int K_  = 1024;   // M_ + S_

// ---- workspace layout (bytes) ----
constexpr size_t SZ_QU  = (size_t)B_ * NH_ * S_ * DH_ * 2;  // 8 MB
constexpr size_t SZ_KN  = (size_t)B_ * NH_ * K_ * DH_ * 2;  // 16 MB
constexpr size_t SZ_VT  = (size_t)B_ * NH_ * DH_ * K_ * 2;  // 16 MB
constexpr size_t SZ_RN  = (size_t)NH_ * K_ * DH_ * 2;       // 2 MB
constexpr size_t SZ_CTX = (size_t)B_ * S_ * H_ * 2;         // 8 MB
constexpr size_t SZ_F32 = (size_t)B_ * S_ * H_ * 4;         // 16 MB
constexpr size_t SZ_FF1 = (size_t)B_ * S_ * FF_ * 2;        // 32 MB

constexpr size_t OFF_QU   = 0;
constexpr size_t OFF_QV   = OFF_QU  + SZ_QU;
constexpr size_t OFF_KN   = OFF_QV  + SZ_QU;
constexpr size_t OFF_VT   = OFF_KN  + SZ_KN;
constexpr size_t OFF_RN   = OFF_VT  + SZ_VT;
constexpr size_t OFF_CTX  = OFF_RN  + SZ_RN;
constexpr size_t OFF_AO   = OFF_CTX + SZ_CTX;   // attn_out f32
constexpr size_t OFF_H    = OFF_AO  + SZ_F32;   // h f32
constexpr size_t OFF_FF1  = OFF_H   + SZ_F32;   // ff1 bf16
constexpr size_t OFF_FF2  = OFF_FF1 + SZ_FF1;   // ff2 f32

__device__ __forceinline__ v8f zero8() {
  v8f z;
#pragma unroll
  for (int i = 0; i < 8; ++i) z[i] = 0.0f;
  return z;
}

__device__ __forceinline__ v8f wmma_bf16(v16bf a, v16bf b, v8f c) {
  return __builtin_amdgcn_wmma_f32_16x16x32_bf16(false, a, false, b,
                                                 (short)0, c, false, false);
}

// A-fragment (16x32): lane's elements are two contiguous 8-element chunks:
//   e in [0,8):  K = 8*half + e        -> rowbase + 8*half
//   e in [8,16): K = 16 + 8*half + e-8 -> rowbase + 16 + 8*half
__device__ __forceinline__ v16bf load_a_frag(const bf16_t* rowbase, int half) {
  v8bf lo = *(const v8bf*)(rowbase + 8 * half);
  v8bf hi = *(const v8bf*)(rowbase + 16 + 8 * half);
  return __builtin_shufflevector(lo, hi, 0, 1, 2, 3, 4, 5, 6, 7, 8, 9, 10, 11,
                                 12, 13, 14, 15);
}
// B-fragment (32x16): lane's elements are two contiguous 8-element chunks:
//   e in [0,8):  K = 8*half + e        -> colbase + 8*half
//   e in [8,16): K = 16 + 8*half + e-8 -> colbase + 16 + 8*half
__device__ __forceinline__ v16bf load_b_frag(const bf16_t* colbase, int half) {
  v8bf lo = *(const v8bf*)(colbase + 8 * half);
  v8bf hi = *(const v8bf*)(colbase + 16 + 8 * half);
  return __builtin_shufflevector(lo, hi, 0, 1, 2, 3, 4, 5, 6, 7, 8, 9, 10, 11,
                                 12, 13, 14, 15);
}

// gfx1250 async DMA: 16B global -> LDS, tracked by ASYNCcnt (ISA 15.18.3 #98)
__device__ __forceinline__ void async_g2l_b128(const void* g, void* lds) {
  unsigned l = (unsigned)(uintptr_t)lds;              // LDS offset = ADDR[31:0]
  unsigned long long ga = (unsigned long long)(uintptr_t)g;
  asm volatile("global_load_async_to_lds_b128 %0, %1, off"
               :: "v"(l), "v"(ga) : "memory");
}
__device__ __forceinline__ void wait_asynccnt0() {
  asm volatile("s_wait_asynccnt 0x0" ::: "memory");
}

// ---------------------------------------------------------------------------
// Generic WMMA GEMM: C[Mrows x N] = A[Mrows x Kd] @ W[Kd x N]  (+ bias, +mode)
// Block 256 = 8 waves; tile 128x64; BK=32; wave tile 32x32 (2x2 WMMA tiles).
// LDS: A row-major [128][LDA], B transposed [64][LDB] so fragments are b128.
// ---------------------------------------------------------------------------
enum { A_F32 = 0, A_CAT = 1, A_BF16 = 2 };
enum { OM_QUQV = 0, OM_KN = 1, OM_VT = 2, OM_RN = 3, OM_F32 = 4, OM_RELU = 5 };

constexpr int LDA = 40;  // padded strides (bank spread); row = 80 B
constexpr int LDB = 40;

template <int AMODE, int OMODE>
__global__ __launch_bounds__(256) void gemm_wmma(
    const float* __restrict__ Af, const bf16_t* __restrict__ Ab,
    const float* __restrict__ Xtok, const float* __restrict__ Mems,
    const float* __restrict__ W, const float* __restrict__ bias,
    const float* __restrict__ bias2, float* __restrict__ outF,
    bf16_t* __restrict__ out0, bf16_t* __restrict__ out1, int Mrows, int N,
    int Kd) {
  __shared__ bf16_t As[128 * LDA];
  __shared__ bf16_t BsT[64 * LDB];

  const int tid  = threadIdx.x;
  const int lane = tid & 31;
  const int half = lane >> 4;
  const int wave = tid >> 5;
  const int wm   = (wave >> 1) * 32;
  const int wn   = (wave & 1) * 32;
  const int bm   = blockIdx.x * 128;
  const int bn   = blockIdx.y * 64;

  v8f acc[2][2];
#pragma unroll
  for (int tm = 0; tm < 2; ++tm)
#pragma unroll
    for (int tn = 0; tn < 2; ++tn) acc[tm][tn] = zero8();

  for (int k0 = 0; k0 < Kd; k0 += 32) {
    // ---- stage A tile (128x32 bf16) ----
    if (AMODE == A_BF16) {
      // async DMA: 512 x 16B chunks, 2 per thread; no VGPR round-trip
#pragma unroll
      for (int i = 0; i < 2; ++i) {
        int idx = tid + i * 256;         // 0..511
        int r = idx >> 2, ch = idx & 3;  // row, 16B chunk (8 bf16)
        async_g2l_b128(Ab + (size_t)(bm + r) * Kd + k0 + ch * 8,
                       &As[r * LDA + ch * 8]);
      }
    } else {
      // f32 -> bf16: float4 loads, packed 8B LDS stores
#pragma unroll
      for (int i = 0; i < 4; ++i) {
        int idx = tid + i * 256;              // 0..1023 chunks of 4
        int r = idx >> 3, c4 = (idx & 7) * 4;
        int gr = bm + r, gk = k0 + c4;
        v4f v;
        if (AMODE == A_CAT) {
          int bb = gr / K_;
          int t  = gr - bb * K_;
          const float* src = (t < M_)
              ? &Mems[((size_t)bb * M_ + t) * H_ + gk]
              : &Xtok[((size_t)bb * S_ + (t - M_)) * H_ + gk];
          v = *(const v4f*)src;
        } else {
          v = *(const v4f*)&Af[(size_t)gr * Kd + gk];
        }
        v4bf o;
#pragma unroll
        for (int e = 0; e < 4; ++e) o[e] = (bf16_t)v[e];
        *(v4bf*)&As[r * LDA + c4] = o;
      }
    }
    // ---- stage B tile (32x64) transposed -> BsT[col][k] ----
#pragma unroll
    for (int i = 0; i < 2; ++i) {
      int idx = tid + i * 256;               // 0..511 chunks of 4
      int r = idx >> 4, c4 = (idx & 15) * 4;
      v4f v = *(const v4f*)&W[(size_t)(k0 + r) * N + (bn + c4)];
#pragma unroll
      for (int e = 0; e < 4; ++e) BsT[(c4 + e) * LDB + r] = (bf16_t)v[e];
    }
    if (k0 + 32 < Kd)  // gfx1250 global_prefetch_b8 of next W tile
      __builtin_prefetch(&W[(size_t)(k0 + 32) * N + bn + (tid & 63)], 0, 1);
    if (AMODE == A_BF16) wait_asynccnt0();
    __syncthreads();

    v16bf af[2], bfr[2];
#pragma unroll
    for (int tm = 0; tm < 2; ++tm)
      af[tm] = load_a_frag(&As[(wm + tm * 16 + (lane & 15)) * LDA], half);
#pragma unroll
    for (int tn = 0; tn < 2; ++tn)
      bfr[tn] = load_b_frag(&BsT[(wn + tn * 16 + (lane & 15)) * LDB], half);
#pragma unroll
    for (int tm = 0; tm < 2; ++tm)
#pragma unroll
      for (int tn = 0; tn < 2; ++tn)
        acc[tm][tn] = wmma_bf16(af[tm], bfr[tn], acc[tm][tn]);
    __syncthreads();
  }

  // ---- writeback (per-mode scatter) ----
#pragma unroll
  for (int tm = 0; tm < 2; ++tm)
#pragma unroll
    for (int tn = 0; tn < 2; ++tn)
#pragma unroll
      for (int g = 0; g < 8; ++g) {
        int m  = bm + wm + tm * 16 + g + (half << 3);
        int nn = bn + wn + tn * 16 + (lane & 15);
        float v = acc[tm][tn][g];
        if (OMODE == OM_QUQV) {
          int bb = m / S_, s = m - bb * S_;
          int hh = nn / DH_, d = nn - hh * DH_;
          size_t o = (((size_t)bb * NH_ + hh) * S_ + s) * DH_ + d;
          out0[o] = (bf16_t)(v + bias[nn]);   // + u_bias
          out1[o] = (bf16_t)(v + bias2[nn]);  // + v_bias
        } else if (OMODE == OM_KN) {  // K: [b][n][j][d] (d contiguous)
          int bb = m / K_, j = m - bb * K_;
          int hh = nn / DH_, d = nn - hh * DH_;
          out0[(((size_t)bb * NH_ + hh) * K_ + j) * DH_ + d] = (bf16_t)v;
        } else if (OMODE == OM_VT) {  // V^T: [b][n][d][j] (j contiguous)
          int bb = m / K_, j = m - bb * K_;
          int hh = nn / DH_, d = nn - hh * DH_;
          out0[(((size_t)bb * NH_ + hh) * DH_ + d) * K_ + j] = (bf16_t)v;
        } else if (OMODE == OM_RN) {  // R: [n][k][d] (d contiguous)
          int hh = nn / DH_, d = nn - hh * DH_;
          out0[((size_t)hh * K_ + m) * DH_ + d] = (bf16_t)v;
        } else if (OMODE == OM_RELU) {
          float t = v + bias[nn];
          out0[(size_t)m * N + nn] = (bf16_t)(t > 0.0f ? t : 0.0f);
        } else {  // OM_F32
          float t = v + (bias ? bias[nn] : 0.0f);
          outF[(size_t)m * N + nn] = t;
        }
      }
}

// ---------------------------------------------------------------------------
// Flash attention with Transformer-XL relative shift.
// Grid (S/64, NH, B), block 128 (4 waves); wave = 16 query rows, 32-key tiles.
// BD closed form: BD[i,j] = BD0[i, j + S-1-i]; 48-col band via WMMA into LDS,
// read back at column (n + 15 - m), independent of j0.
// ---------------------------------------------------------------------------
constexpr int PSTR = 40;  // pbuf row stride

__global__ __launch_bounds__(128) void attn_flash(
    const bf16_t* __restrict__ Qu, const bf16_t* __restrict__ Qv,
    const bf16_t* __restrict__ Kn, const bf16_t* __restrict__ Vt,
    const bf16_t* __restrict__ Rn, bf16_t* __restrict__ Ctx) {
  const int tid = threadIdx.x, lane = tid & 31, w = tid >> 5;
  const int half = lane >> 4;
  const int b = blockIdx.z, h = blockIdx.y;
  const int i0 = blockIdx.x * 64 + w * 16;

  __shared__ float  bdb[4][16 * 48];    // per-wave BD0 band
  __shared__ bf16_t pbuf[4][16 * PSTR]; // per-wave P tile (A-frag relayout)

  const size_t headQ = ((size_t)b * NH_ + h) * S_ * DH_;
  const size_t headK = ((size_t)b * NH_ + h) * K_ * DH_;  // [j][d]
  const size_t headV = ((size_t)b * NH_ + h) * DH_ * K_;  // [d][j]
  const size_t headR = (size_t)h * K_ * DH_;              // [k][d]

  // Q fragments (u/v bias pre-folded): two 16x32 A-fragments each
  v16bf qu[2], qv[2];
  {
    const bf16_t* qrow = Qu + headQ + (size_t)(i0 + (lane & 15)) * DH_;
    const bf16_t* vrow = Qv + headQ + (size_t)(i0 + (lane & 15)) * DH_;
#pragma unroll
    for (int t = 0; t < 2; ++t) {
      qu[t] = load_a_frag(qrow + t * 32, half);
      qv[t] = load_a_frag(vrow + t * 32, half);
    }
  }

  v8f o[4];
  float rmax[8], rsum[8];
#pragma unroll
  for (int t = 0; t < 4; ++t) o[t] = zero8();
#pragma unroll
  for (int g = 0; g < 8; ++g) { rmax[g] = -3.0e38f; rsum[g] = 0.0f; }

  const int jend = (K_ < i0 + M_ + 16) ? K_ : (i0 + M_ + 16);
  for (int j0 = 0; j0 < jend; j0 += 32) {
    // ---------- AC: (q+u) . k^T, two 16x16 output tiles ----------
    v8f sc0 = zero8(), sc1 = zero8();
    {
      const bf16_t* c0 = Kn + headK + (size_t)(j0 + (lane & 15)) * DH_;
      const bf16_t* c1 = c0 + 16 * DH_;
#pragma unroll
      for (int t = 0; t < 2; ++t) {
        sc0 = wmma_bf16(qu[t], load_b_frag(c0 + t * 32, half), sc0);
        sc1 = wmma_bf16(qu[t], load_b_frag(c1 + t * 32, half), sc1);
      }
    }

    // ---------- BD band: 48 columns starting at jb ----------
    const int jb = j0 + (S_ - 1) - i0 - 15;
#pragma unroll
    for (int bt = 0; bt < 3; ++bt) {
      v8f bda = zero8();
      int col = jb + bt * 16 + (lane & 15);
      int colc = col < 0 ? 0 : (col > K_ - 1 ? K_ - 1 : col);
      const bf16_t* cr = Rn + headR + (size_t)colc * DH_;
#pragma unroll
      for (int t = 0; t < 2; ++t)
        bda = wmma_bf16(qv[t], load_b_frag(cr + t * 32, half), bda);
#pragma unroll
      for (int g = 0; g < 8; ++g) {
        int m = g + (half << 3);
        bdb[w][m * 48 + bt * 16 + (lane & 15)] = bda[g];
      }
    }

    // ---------- combine, mask, online softmax ----------
#pragma unroll
    for (int g = 0; g < 8; ++g) {
      int m = g + (half << 3);
      int n2a = (lane & 15);
      int n2b = 16 + (lane & 15);
      float s0 = (sc0[g] + bdb[w][m * 48 + (n2a + 15 - m)]) * 0.125f;
      float s1 = (sc1[g] + bdb[w][m * 48 + (n2b + 15 - m)]) * 0.125f;
      if (j0 + n2a > i0 + m + M_) s0 = -1.0e30f;
      if (j0 + n2b > i0 + m + M_) s1 = -1.0e30f;
      float mx = fmaxf(s0, s1);
#pragma unroll
      for (int off = 1; off < 16; off <<= 1)
        mx = fmaxf(mx, __shfl_xor(mx, off, 16));
      float nm = fmaxf(rmax[g], mx);
      float c  = __expf(rmax[g] - nm);
      rmax[g]  = nm;
      float p0 = __expf(s0 - nm), p1 = __expf(s1 - nm);
      float ps = p0 + p1;
#pragma unroll
      for (int off = 1; off < 16; off <<= 1) ps += __shfl_xor(ps, off, 16);
      rsum[g] = rsum[g] * c + ps;
#pragma unroll
      for (int t = 0; t < 4; ++t) o[t][g] *= c;
      pbuf[w][m * PSTR + (lane & 15)]      = (bf16_t)p0;
      pbuf[w][m * PSTR + 16 + (lane & 15)] = (bf16_t)p1;
    }

    // ---------- P @ V  (16x32 x 32x64) ----------
    v16bf pf = load_a_frag(&pbuf[w][(lane & 15) * PSTR], half);
#pragma unroll
    for (int t = 0; t < 4; ++t) {
      const bf16_t* cv =
          Vt + headV + (size_t)(t * 16 + (lane & 15)) * K_ + j0;
      o[t] = wmma_bf16(pf, load_b_frag(cv, half), o[t]);
    }
  }

  // ---------- normalize + store ctx (bf16, [B,S,H]) ----------
#pragma unroll
  for (int t = 0; t < 4; ++t)
#pragma unroll
    for (int g = 0; g < 8; ++g) {
      int m = g + (half << 3);
      int d = t * 16 + (lane & 15);
      float v = o[t][g] / rsum[g];
      Ctx[((size_t)b * S_ + (i0 + m)) * H_ + h * DH_ + d] = (bf16_t)v;
    }
}

// ---------------------------------------------------------------------------
// out[row,:] = LayerNorm(A[row,:] + Badd[row,:]) * gamma + beta
// 256 threads/row, contiguous float4 per thread.
// ---------------------------------------------------------------------------
__global__ __launch_bounds__(256) void ln_add_kernel(
    const float* __restrict__ A, const float* __restrict__ Badd,
    const float* __restrict__ gamma, const float* __restrict__ beta,
    float* __restrict__ out) {
  const int row = blockIdx.x, tid = threadIdx.x;
  __shared__ float red[256];
  const int c0 = tid * 4;
  v4f a = *(const v4f*)(A + (size_t)row * H_ + c0);
  v4f b = *(const v4f*)(Badd + (size_t)row * H_ + c0);
  v4f v;
  float s = 0.0f;
#pragma unroll
  for (int i = 0; i < 4; ++i) {
    v[i] = a[i] + b[i];
    s += v[i];
  }
  red[tid] = s;
  __syncthreads();
  for (int o = 128; o > 0; o >>= 1) {
    if (tid < o) red[tid] += red[tid + o];
    __syncthreads();
  }
  float mu = red[0] * (1.0f / H_);
  __syncthreads();
  float s2 = 0.0f;
#pragma unroll
  for (int i = 0; i < 4; ++i) {
    float d = v[i] - mu;
    s2 += d * d;
  }
  red[tid] = s2;
  __syncthreads();
  for (int o = 128; o > 0; o >>= 1) {
    if (tid < o) red[tid] += red[tid + o];
    __syncthreads();
  }
  float rstd = rsqrtf(red[0] * (1.0f / H_) + 1e-5f);
  v4f gm = *(const v4f*)(gamma + c0);
  v4f bt = *(const v4f*)(beta + c0);
  v4f r;
#pragma unroll
  for (int i = 0; i < 4; ++i) r[i] = (v[i] - mu) * rstd * gm[i] + bt[i];
  *(v4f*)(out + (size_t)row * H_ + c0) = r;
}

// ---------------------------------------------------------------------------
extern "C" void kernel_launch(void* const* d_in, const int* in_sizes, int n_in,
                              void* d_out, int out_size, void* d_ws,
                              size_t ws_size, hipStream_t stream) {
  const float* x     = (const float*)d_in[0];   // [B,S,H]
  const float* rel   = (const float*)d_in[1];   // [K,H]
  const float* mems  = (const float*)d_in[2];   // [B,M,H]
  const float* Wq    = (const float*)d_in[5];
  const float* Wk    = (const float*)d_in[6];
  const float* Wv    = (const float*)d_in[7];
  const float* Wr    = (const float*)d_in[8];
  const float* Wo    = (const float*)d_in[9];
  const float* ub    = (const float*)d_in[10];  // [NH,DH] == [H]
  const float* vbias = (const float*)d_in[11];
  const float* gamma = (const float*)d_in[12];
  const float* beta  = (const float*)d_in[13];
  const float* W1    = (const float*)d_in[14];
  const float* b1    = (const float*)d_in[15];
  const float* W2    = (const float*)d_in[16];
  const float* b2    = (const float*)d_in[17];

  char* ws = (char*)d_ws;
  bf16_t* Qu   = (bf16_t*)(ws + OFF_QU);
  bf16_t* Qv   = (bf16_t*)(ws + OFF_QV);
  bf16_t* Kn   = (bf16_t*)(ws + OFF_KN);
  bf16_t* Vt   = (bf16_t*)(ws + OFF_VT);
  bf16_t* Rn   = (bf16_t*)(ws + OFF_RN);
  bf16_t* Ctx  = (bf16_t*)(ws + OFF_CTX);
  float*  AO   = (float*)(ws + OFF_AO);
  float*  Hbuf = (float*)(ws + OFF_H);
  bf16_t* FF1  = (bf16_t*)(ws + OFF_FF1);
  float*  FF2  = (float*)(ws + OFF_FF2);
  float*  y    = (float*)d_out;

  const int BS = B_ * S_;  // 4096
  const int BK = B_ * K_;  // 8192

  // 1) q -> Qu/Qv (u/v bias folded, head-major bf16)
  gemm_wmma<A_F32, OM_QUQV><<<dim3(BS / 128, H_ / 64), 256, 0, stream>>>(
      x, nullptr, nullptr, nullptr, Wq, ub, vbias, nullptr, Qu, Qv, BS, H_, H_);
  // 2) k = cat @ Wk -> Kn[b][n][j][d]
  gemm_wmma<A_CAT, OM_KN><<<dim3(BK / 128, H_ / 64), 256, 0, stream>>>(
      nullptr, nullptr, x, mems, Wk, nullptr, nullptr, nullptr, Kn, nullptr,
      BK, H_, H_);
  // 3) v = cat @ Wv -> Vt[b][n][d][j]
  gemm_wmma<A_CAT, OM_VT><<<dim3(BK / 128, H_ / 64), 256, 0, stream>>>(
      nullptr, nullptr, x, mems, Wv, nullptr, nullptr, nullptr, Vt, nullptr,
      BK, H_, H_);
  // 4) r = rel @ Wr -> Rn[n][k][d]
  gemm_wmma<A_F32, OM_RN><<<dim3(K_ / 128, H_ / 64), 256, 0, stream>>>(
      rel, nullptr, nullptr, nullptr, Wr, nullptr, nullptr, nullptr, Rn,
      nullptr, K_, H_, H_);
  // 5) flash attention -> Ctx bf16
  attn_flash<<<dim3(S_ / 64, NH_, B_), 128, 0, stream>>>(Qu, Qv, Kn, Vt, Rn,
                                                         Ctx);
  // 6) attn_out = ctx @ Wo (f32), ctx staged via async DMA
  gemm_wmma<A_BF16, OM_F32><<<dim3(BS / 128, H_ / 64), 256, 0, stream>>>(
      nullptr, Ctx, nullptr, nullptr, Wo, nullptr, nullptr, AO, nullptr,
      nullptr, BS, H_, H_);
  // 7) h = LN(attn_out + x)
  ln_add_kernel<<<dim3(BS), 256, 0, stream>>>(AO, x, gamma, beta, Hbuf);
  // 8) ff1 = relu(h @ W1 + b1) (bf16)
  gemm_wmma<A_F32, OM_RELU><<<dim3(BS / 128, FF_ / 64), 256, 0, stream>>>(
      Hbuf, nullptr, nullptr, nullptr, W1, b1, nullptr, nullptr, FF1, nullptr,
      BS, FF_, H_);
  // 9) ff2 = ff1 @ W2 + b2 (f32), ff1 staged via async DMA
  gemm_wmma<A_BF16, OM_F32><<<dim3(BS / 128, H_ / 64), 256, 0, stream>>>(
      nullptr, FF1, nullptr, nullptr, W2, b2, nullptr, FF2, nullptr, nullptr,
      BS, H_, FF_);
  // 10) y = LN(ff2 + h)
  ln_add_kernel<<<dim3(BS), 256, 0, stream>>>(FF2, Hbuf, gamma, beta, y);
  // 11) new_mems = cat[:, -M:] == token_embedding (since M == S)
  hipMemcpyAsync(y + (size_t)BS * H_, x, (size_t)BS * H_ * sizeof(float),
                 hipMemcpyDeviceToDevice, stream);
}
